// HybridAutoencoder_65481071396601
// MI455X (gfx1250) — matrix operand order repaired and checked by reading, not compiled, not run on hardware
//
#include <hip/hip_runtime.h>
#include <hip/hip_bf16.h>

#define B_ROWS 16384
#define IN_DIM 4096
#define H1D    128
#define H2D    64
#define LATENT 32
#define REPS   3

typedef __bf16 bf16_t;
typedef __attribute__((ext_vector_type(16))) bf16_t v16bf;
typedef __attribute__((ext_vector_type(8)))  float  v8f;

// ---- LDS pool overlay (bytes). Pool = 48 KB. ----
// Phase1 live: AS0/AS1 (f32, double-buffered), BS0/BS1 (bf16, double-buffered), HS(out)
// Phase2 live: HS(in), W2T, H2S(out)
// Phase3 live: H2S(in), W3T, QS(out)
// Phase4 live: QS(in), WDT
#define OFF_HS    0        // 64 x 128 bf16 = 16384 B
#define OFF_AS0   16384    // 64 x 32  f32  =  8192 B
#define OFF_AS1   24576    // 64 x 32  f32  =  8192 B
#define OFF_BS0   32768    // 128 x 32 bf16 =  8192 B
#define OFF_BS1   40960    // 128 x 32 bf16 =  8192 B
#define OFF_W2T   16384    // 64 x 128 bf16 = 16384 B  (reuses AS0+AS1)
#define OFF_H2S   32768    // 64 x 64  bf16 =  8192 B  (reuses BS0)
#define OFF_W3T   0        // 32 x 64  bf16 =  4096 B  (reuses HS)
#define OFF_QS    4096     // 64 x 32  bf16 =  4096 B
#define OFF_WDT   8192     // 512 x 32 bf16 = 32768 B  (reuses rest)
#define LDS_BYTES 49152

// d_ws layout: W1t bf16 [128][4096] then Wdt bf16 [4096][32]
#define WS_W1T_BYTES (H1D * IN_DIM * 2)       // 1048576
#define WS_WDT_OFF   WS_W1T_BYTES
#define WS_WDT_BYTES (IN_DIM * LATENT * 2)    // 262144

union FragBF {
  uint4  q[2];
  v16bf  v;
};
union F32x4 {
  uint4 u;
  float f[4];
};

__device__ __forceinline__ v16bf load_frag(const char* lds, int byte0, int byte1) {
  FragBF f;
  f.q[0] = *(const uint4*)(lds + byte0);
  f.q[1] = *(const uint4*)(lds + byte1);
  return f.v;
}

// Build a bf16 A-fragment from f32 data resident in LDS (convert-on-read).
__device__ __forceinline__ v16bf load_frag_f32(const char* lds, int b0, int b1) {
  F32x4 q0, q1, q2, q3;
  q0.u = *(const uint4*)(lds + b0);
  q1.u = *(const uint4*)(lds + b0 + 16);
  q2.u = *(const uint4*)(lds + b1);
  q3.u = *(const uint4*)(lds + b1 + 16);
  v16bf a;
#pragma unroll
  for (int i = 0; i < 4; ++i) {
    a[i]      = (bf16_t)q0.f[i];
    a[4 + i]  = (bf16_t)q1.f[i];
    a[8 + i]  = (bf16_t)q2.f[i];
    a[12 + i] = (bf16_t)q3.f[i];
  }
  return a;
}

__device__ __forceinline__ v8f v8f_splat(float x) {
  v8f r;
#pragma unroll
  for (int i = 0; i < 8; ++i) r[i] = x;
  return r;
}

__device__ __forceinline__ v8f wmma_bf16(v16bf a, v16bf b, v8f c) {
  return __builtin_amdgcn_wmma_f32_16x16x32_bf16(
      false, a, false, b, (short)0, c, false, false);
}

// Async copy of 16 bytes: global (saddr + 32-bit voffset) -> LDS byte address.
// Tracked with ASYNCcnt; completion via s_wait_asynccnt + workgroup barrier.
__device__ __forceinline__ void async_b128(unsigned lds_addr, int goff, const void* base) {
  asm volatile("global_load_async_to_lds_b128 %0, %1, %2"
               :: "v"(lds_addr), "v"(goff), "s"(base)
               : "memory");
}
__device__ __forceinline__ void async_wait0() {
  asm volatile("s_wait_asynccnt 0x0" ::: "memory");
}

// ------------- One-time weight prep: bf16 + transpose into d_ws -------------
__global__ __launch_bounds__(256)
void prep_w1t(const float* __restrict__ W1, bf16_t* __restrict__ W1t) {
  int idx = blockIdx.x * 256 + threadIdx.x;    // over 4096*128
  int k = idx >> 7, n = idx & 127;
  W1t[(size_t)n * IN_DIM + k] = (bf16_t)W1[idx];
}
__global__ __launch_bounds__(256)
void prep_wdt(const float* __restrict__ Wd, bf16_t* __restrict__ Wdt) {
  int idx = blockIdx.x * 256 + threadIdx.x;    // over 32*4096
  int k = idx >> 12, j = idx & 4095;
  Wdt[(size_t)j * LATENT + k] = (bf16_t)Wd[idx];
}

// ---------------------------- Fused main kernel ----------------------------
__global__ __launch_bounds__(128)
void fused_hybrid_autoencoder(const float* __restrict__ x,
                              const bf16_t* __restrict__ W1t, const float* __restrict__ b1,
                              const float* __restrict__ W2,   const float* __restrict__ b2,
                              const float* __restrict__ W3,   const float* __restrict__ b3,
                              const float* __restrict__ qp,
                              const bf16_t* __restrict__ Wdt, const float* __restrict__ bd,
                              float* __restrict__ out)
{
  __shared__ __align__(16) char lds[LDS_BYTES];
  const unsigned ldsBase = (unsigned)(uintptr_t)lds;

  const int tid  = threadIdx.x;
  const int lane = tid & 31;
  const int wv   = tid >> 5;        // wave 0..3 owns rows wv*16..wv*16+15
  const int nl   = lane & 15;
  const int hi   = lane >> 4;
  const int aOff0 = hi * 32;        // bf16 A frag: e[0..7]  K = 16*hi+0..7
  const int aOff1 = hi * 32 + 16;   //             e[8..15] K = 16*hi+8..15
  const int bOff  = hi * 32;        // B frag: K = 16*hi+0..15, 32 contiguous B

  const int r0 = blockIdx.x * 64;

  // ===== Phase 1: h1 = relu(x @ W1 + b1), double-buffered async staging ====
  v8f acc1[8];
#pragma unroll
  for (int nt = 0; nt < 8; ++nt) acc1[nt] = v8f_splat(0.f);

  const int aRowByte = (wv * 16 + nl) * 128;    // AS row = 32 f32 = 128 B

  // lane-op indices for the cooperative async stages
  const int xRow = tid >> 3, xKq = tid & 7;     // + t*16 rows per t
  const int wN = tid >> 2, wPart = tid & 3;     // + t*32 rows per t

  // prologue: stage chunk 0 into buffer 0
#pragma unroll
  for (int t = 0; t < 4; ++t) {
    int row = xRow + t * 16;
    async_b128(ldsBase + OFF_AS0 + row * 128 + xKq * 16,
               (r0 + row) * (IN_DIM * 4) + xKq * 16, x);
  }
#pragma unroll
  for (int t = 0; t < 4; ++t) {
    int n = wN + t * 32;
    async_b128(ldsBase + OFF_BS0 + n * 64 + wPart * 16,
               n * (IN_DIM * 2) + wPart * 16, W1t);
  }
  async_wait0();
  __syncthreads();

  for (int kc = 0; kc < IN_DIM / 32; ++kc) {
    const int abuf = (kc & 1) ? OFF_AS1 : OFF_AS0;
    const int bbuf = (kc & 1) ? OFF_BS1 : OFF_BS0;
    // prefetch chunk kc+1 into the other buffer
    if (kc + 1 < IN_DIM / 32) {
      const int abufN = (kc & 1) ? OFF_AS0 : OFF_AS1;
      const int bbufN = (kc & 1) ? OFF_BS0 : OFF_BS1;
      const int k0n = (kc + 1) * 32;
#pragma unroll
      for (int t = 0; t < 4; ++t) {
        int row = xRow + t * 16;
        async_b128(ldsBase + abufN + row * 128 + xKq * 16,
                   (r0 + row) * (IN_DIM * 4) + k0n * 4 + xKq * 16, x);
      }
#pragma unroll
      for (int t = 0; t < 4; ++t) {
        int n = wN + t * 32;
        async_b128(ldsBase + bbufN + n * 64 + wPart * 16,
                   n * (IN_DIM * 2) + k0n * 2 + wPart * 16, W1t);
      }
    }
    // compute on current buffer
    v16bf a = load_frag_f32(lds, abuf + aRowByte + hi * 64,
                                 abuf + aRowByte + hi * 64 + 32);
#pragma unroll
    for (int nt = 0; nt < 8; ++nt) {
      int brow = bbuf + (nt * 16 + nl) * 64;
      v16bf b = load_frag(lds, brow + bOff, brow + bOff + 16);
      acc1[nt] = wmma_bf16(a, b, acc1[nt]);
    }
    async_wait0();      // my prefetch landed
    __syncthreads();    // everyone's reads of current buffer done
  }

  // bias + relu -> HS (64 x 128 bf16), then stage W2T (reuses AS regions)
  {
    bf16_t* Hs = (bf16_t*)(lds + OFF_HS);
#pragma unroll
    for (int nt = 0; nt < 8; ++nt) {
      int n = nt * 16 + nl;
      float bn = b1[n];
#pragma unroll
      for (int r = 0; r < 8; ++r) {
        float hv = acc1[nt][r] + bn;
        hv = hv > 0.f ? hv : 0.f;
        Hs[(wv * 16 + hi * 8 + r) * 128 + n] = (bf16_t)hv;
      }
    }
  }
  // ===== Phase 2: h2 = relu(h1 @ W2 + b2) [64 x 64] =====
  for (int i = tid; i < 128 * 16; i += 128) {
    int k = i >> 4, nq = i & 15;
    float4 v = *(const float4*)(W2 + (size_t)k * H2D + nq * 4);
    bf16_t* base = (bf16_t*)(lds + OFF_W2T);   // W2T[n][k], stride 128
    base[(nq * 4 + 0) * 128 + k] = (bf16_t)v.x;
    base[(nq * 4 + 1) * 128 + k] = (bf16_t)v.y;
    base[(nq * 4 + 2) * 128 + k] = (bf16_t)v.z;
    base[(nq * 4 + 3) * 128 + k] = (bf16_t)v.w;
  }
  __syncthreads();

  v8f acc2[4];
#pragma unroll
  for (int nt = 0; nt < 4; ++nt) acc2[nt] = v8f_splat(0.f);
  {
    int hRow = OFF_HS + (wv * 16 + nl) * 256;
#pragma unroll
    for (int k0 = 0; k0 < 128; k0 += 32) {
      v16bf a = load_frag(lds, hRow + k0 * 2 + aOff0, hRow + k0 * 2 + aOff1);
#pragma unroll
      for (int nt = 0; nt < 4; ++nt) {
        int brow = OFF_W2T + (nt * 16 + nl) * 256;
        v16bf b = load_frag(lds, brow + k0 * 2 + bOff, brow + k0 * 2 + bOff + 16);
        acc2[nt] = wmma_bf16(a, b, acc2[nt]);
      }
    }
  }
  {
    bf16_t* H2s = (bf16_t*)(lds + OFF_H2S);
#pragma unroll
    for (int nt = 0; nt < 4; ++nt) {
      int n = nt * 16 + nl;
      float bn = b2[n];
#pragma unroll
      for (int r = 0; r < 8; ++r) {
        float hv = acc2[nt][r] + bn;
        hv = hv > 0.f ? hv : 0.f;
        H2s[(wv * 16 + hi * 8 + r) * 64 + n] = (bf16_t)hv;
      }
    }
  }
  __syncthreads();   // HS reads done -> safe to overwrite with W3T

  // ===== Phase 3: q = cos(h2 @ W3 + b3 + sum(qparams)) =====
  for (int i = tid; i < 64 * 8; i += 128) {
    int k = i >> 3, nq = i & 7;
    float4 v = *(const float4*)(W3 + (size_t)k * LATENT + nq * 4);
    bf16_t* base = (bf16_t*)(lds + OFF_W3T);   // W3T[n][k], stride 64
    base[(nq * 4 + 0) * 64 + k] = (bf16_t)v.x;
    base[(nq * 4 + 1) * 64 + k] = (bf16_t)v.y;
    base[(nq * 4 + 2) * 64 + k] = (bf16_t)v.z;
    base[(nq * 4 + 3) * 64 + k] = (bf16_t)v.w;
  }
  __syncthreads();

  v8f acc3[2];
#pragma unroll
  for (int nt = 0; nt < 2; ++nt) acc3[nt] = v8f_splat(0.f);
  {
    int h2Row = OFF_H2S + (wv * 16 + nl) * 128;
#pragma unroll
    for (int k0 = 0; k0 < 64; k0 += 32) {
      v16bf a = load_frag(lds, h2Row + k0 * 2 + aOff0, h2Row + k0 * 2 + aOff1);
#pragma unroll
      for (int nt = 0; nt < 2; ++nt) {
        int brow = OFF_W3T + (nt * 16 + nl) * 128;
        v16bf b = load_frag(lds, brow + k0 * 2 + bOff, brow + k0 * 2 + bOff + 16);
        acc3[nt] = wmma_bf16(a, b, acc3[nt]);
      }
    }
  }
  {
    bf16_t* Qs = (bf16_t*)(lds + OFF_QS);      // Qs[m][l], stride 32
#pragma unroll
    for (int nt = 0; nt < 2; ++nt) {
      int n = nt * 16 + nl;
      float offn = b3[n];
#pragma unroll
      for (int r6 = 0; r6 < 2 * REPS; ++r6) offn += qp[r6 * LATENT + n];
#pragma unroll
      for (int r = 0; r < 8; ++r) {
        float qv = cosf(acc3[nt][r] + offn);
        Qs[(wv * 16 + hi * 8 + r) * 32 + n] = (bf16_t)qv;
      }
    }
  }
  __syncthreads();   // H2S reads done -> safe to overwrite with WDT

  // ===== Phase 4: out = q @ Wd + bd (K = LATENT = 32, single WMMA step) ====
  const int qRow = OFF_QS + (wv * 16 + nl) * 64;
  v16bf aq = load_frag(lds, qRow + aOff0, qRow + aOff1);

  const int dJ = tid >> 2, dPart = tid & 3;    // + t*32 rows per t
  for (int j0 = 0; j0 < IN_DIM; j0 += 512) {
    __syncthreads();   // previous chunk's WDT reads complete
#pragma unroll
    for (int t = 0; t < 16; ++t) {
      int j = dJ + t * 32;
      async_b128(ldsBase + OFF_WDT + j * 64 + dPart * 16,
                 (j0 + j) * (LATENT * 2) + dPart * 16, Wdt);
    }
    async_wait0();
    __syncthreads();
#pragma unroll 4
    for (int nt = 0; nt < 32; ++nt) {
      int n = j0 + nt * 16 + nl;
      v8f c = v8f_splat(bd[n]);                // bias preloaded into C
      int brow = OFF_WDT + (nt * 16 + nl) * 64;
      v16bf b = load_frag(lds, brow + bOff, brow + bOff + 16);
      c = wmma_bf16(aq, b, c);
      float* orow = out + (size_t)(r0 + wv * 16 + hi * 8) * IN_DIM + n;
#pragma unroll
      for (int r = 0; r < 8; ++r) orow[(size_t)r * IN_DIM] = c[r];
    }
  }
}

extern "C" void kernel_launch(void* const* d_in, const int* in_sizes, int n_in,
                              void* d_out, int out_size, void* d_ws, size_t ws_size,
                              hipStream_t stream) {
  const float* x  = (const float*)d_in[0];
  const float* W1 = (const float*)d_in[1];
  const float* b1 = (const float*)d_in[2];
  const float* W2 = (const float*)d_in[3];
  const float* b2 = (const float*)d_in[4];
  const float* W3 = (const float*)d_in[5];
  const float* b3 = (const float*)d_in[6];
  const float* qp = (const float*)d_in[7];
  const float* Wd = (const float*)d_in[8];
  const float* bd = (const float*)d_in[9];
  float* out = (float*)d_out;
  (void)in_sizes; (void)n_in; (void)out_size; (void)ws_size;

  bf16_t* W1t = (bf16_t*)d_ws;                               // 1 MB
  bf16_t* Wdt = (bf16_t*)((char*)d_ws + WS_WDT_OFF);         // 256 KB

  // One-time (per launch) weight prep: bf16 + transpose, L2-resident after.
  hipLaunchKernelGGL(prep_w1t, dim3((IN_DIM * H1D) / 256), dim3(256), 0, stream, W1, W1t);
  hipLaunchKernelGGL(prep_wdt, dim3((LATENT * IN_DIM) / 256), dim3(256), 0, stream, Wd, Wdt);

  dim3 grid(B_ROWS / 64);   // 256 blocks, 64 batch rows each
  dim3 block(128);          // 4 waves (wave32)
  hipLaunchKernelGGL(fused_hybrid_autoencoder, grid, block, 0, stream,
                     x, W1t, b1, W2, b2, W3, b3, qp, Wdt, bd, out);
}
